// MultiHeadSelfAttention_5798205849868
// MI455X (gfx1250) — matrix operand order repaired and checked by reading, not compiled, not run on hardware
//
#include <hip/hip_runtime.h>
#include <hip/hip_bf16.h>

// ---------------------------------------------------------------------------
// MultiHeadSelfAttention for MI455X (gfx1250): GroupNorm + QKV + flash-attn +
// proj + residual.  All matmuls on v_wmma_f32_16x16x32_f16 (f16 in, f32 acc).
// Attention K/V tiles are block-staged in LDS (8x L2-traffic reduction).
// ---------------------------------------------------------------------------

typedef _Float16 v16h __attribute__((ext_vector_type(16)));
typedef float    v8f  __attribute__((ext_vector_type(8)));

#define BATCH   4
#define CCH     256
#define TT      4096
#define NHEADS  4
#define CH      64
#define BH      (BATCH * NHEADS)   // 16
#define NGROUPS 32
#define CPG     (CCH / NGROUPS)    // 8
#define EPSV    1e-5f
#define QK_SCALE 0.35355339059327373f   // 64^-0.25
#define LOG2E    1.4426950408889634f
#define LPITCH  72                       // LDS row pitch (f16) for 64-wide tiles

static __device__ __forceinline__ v8f wmma16(v16h a, v16h b, v8f c) {
    return __builtin_amdgcn_wmma_f32_16x16x32_f16(false, a, false, b, (short)0, c, false, false);
}

// A fragment (16x32 f16): lane<16 -> K 0..7 & 16..23 of row M=lane;
// lane>=16 -> K 8..15 & 24..31 of row M=lane-16.  ld in f16 elements.
static __device__ __forceinline__ v16h load_a(const _Float16* base, int ld) {
    const int lane = threadIdx.x & 31;
    const int row  = lane & 15;
    const int sel  = lane >> 4;
    const _Float16* p = base + row * ld + sel * 8;
    union { uint4 u[2]; v16h h; } r;
    r.u[0] = *(const uint4*)(p);
    r.u[1] = *(const uint4*)(p + 16);
    return r.h;
}

// B fragment (32x16 f16): lane<16 -> K 0..7 & 16..23 of column N=lane&15;
// lane>=16 -> K 8..15 & 24..31.  Memory layout: mem[n*ld + k].
static __device__ __forceinline__ v16h load_b(const _Float16* base, int ld) {
    const int lane = threadIdx.x & 31;
    const int col  = lane & 15;
    const int sel  = lane >> 4;
    const _Float16* p = base + col * ld + sel * 8;
    union { uint4 u[2]; v16h h; } r;
    r.u[0] = *(const uint4*)(p);
    r.u[1] = *(const uint4*)(p + 16);
    return r.h;
}

// ---------------- K0: convert weights to f16 ----------------
__global__ void k_cvt_w(const float* qkv_w, const float* proj_w,
                        _Float16* wqkv, _Float16* wproj) {
    int i = blockIdx.x * blockDim.x + threadIdx.x;   // 1024*256 = 262144
    if (i < 768 * 256)       wqkv[i] = (_Float16)qkv_w[i];
    else                     wproj[i - 768 * 256] = (_Float16)proj_w[i - 768 * 256];
}

// ---------------- K1: groupnorm stats (mean, rstd) ----------------
__global__ void k_gn_stats(const float* __restrict__ x, float* __restrict__ stats) {
    __shared__ float ssum[256], ssq[256];
    const int blk = blockIdx.x;          // b*32 + g
    const float* p = x + (size_t)blk * (CPG * TT);
    float s = 0.f, q = 0.f;
    for (int i = threadIdx.x; i < CPG * TT; i += 256) {
        float v = p[i];
        s += v; q += v * v;
    }
    ssum[threadIdx.x] = s; ssq[threadIdx.x] = q;
    __syncthreads();
    for (int off = 128; off > 0; off >>= 1) {
        if (threadIdx.x < off) {
            ssum[threadIdx.x] += ssum[threadIdx.x + off];
            ssq[threadIdx.x]  += ssq[threadIdx.x + off];
        }
        __syncthreads();
    }
    if (threadIdx.x == 0) {
        const float inv_n = 1.f / (float)(CPG * TT);
        float mean = ssum[0] * inv_n;
        float var  = ssq[0] * inv_n - mean * mean;
        stats[blk * 2 + 0] = mean;
        stats[blk * 2 + 1] = rsqrtf(var + EPSV);
    }
}

// ---------------- K2: normalize + transpose -> xnT[b][t][c] f16 ----------------
__global__ void k_gn_apply(const float* __restrict__ x,
                           const float* __restrict__ nw, const float* __restrict__ nb,
                           const float* __restrict__ stats,
                           _Float16* __restrict__ xnT) {
    __shared__ float tile[32][33];
    const int blk = blockIdx.x;                 // b*1024 + ct*128 + tt
    const int b  = blk >> 10;
    const int ct = (blk >> 7) & 7;
    const int tt = blk & 127;
    const int c0 = ct * 32, t0 = tt * 32;
    {
        const int tcol = threadIdx.x & 31;
        const int crow = threadIdx.x >> 5;      // 0..7
        for (int k = 0; k < 4; ++k) {
            const int c = c0 + crow + k * 8;
            tile[crow + k * 8][tcol] = x[((size_t)(b * CCH + c)) * TT + t0 + tcol];
        }
    }
    __syncthreads();
    {
        const int c2   = threadIdx.x & 31;
        const int trow = threadIdx.x >> 5;
        const int c    = c0 + c2;
        const int g    = c >> 3;
        const float mean = stats[(b * NGROUPS + g) * 2 + 0];
        const float rstd = stats[(b * NGROUPS + g) * 2 + 1];
        const float sc = rstd * nw[c];
        const float sh = nb[c] - mean * sc;
        for (int k = 0; k < 4; ++k) {
            const int t = t0 + trow + k * 8;
            float v = tile[c2][trow + k * 8];
            xnT[((size_t)(b * TT + t)) * CCH + c] = (_Float16)(v * sc + sh);
        }
    }
}

// ---------------- K3: QKV GEMM (WMMA) -> qT, kT (scaled, t-major), v ----------
__global__ void k_qkv(const _Float16* __restrict__ wqkv,
                      const float* __restrict__ qkv_b,
                      const _Float16* __restrict__ xnT,
                      _Float16* __restrict__ qT, _Float16* __restrict__ kT,
                      _Float16* __restrict__ vbuf) {
    const int warp = threadIdx.x >> 5;
    const int strip = blockIdx.x * 8 + warp;        // 12288 strips
    const int b   = strip / 3072;
    const int rem = strip % 3072;
    const int o0  = (rem / 64) * 16;                // 0..752
    const int t0  = (rem % 64) * 64;
    const int lane = threadIdx.x & 31;
    const int m   = lane & 15;
    const int sel = lane >> 4;

    v16h af[8];
#pragma unroll
    for (int kc = 0; kc < 8; ++kc)
        af[kc] = load_a(wqkv + o0 * CCH + kc * 32, CCH);

    float bias[8];
#pragma unroll
    for (int i = 0; i < 8; ++i) bias[i] = qkv_b[o0 + sel * 8 + i];

    const int omod = o0 % 192;
    const int sec  = omod / 64;             // 0=q 1=k 2=v
    const int head = o0 / 192;
    const int bh   = b * NHEADS + head;
    const int r0   = (omod - sec * 64) + sel * 8;

#pragma unroll
    for (int j = 0; j < 4; ++j) {
        v8f acc;
#pragma unroll
        for (int i = 0; i < 8; ++i) acc[i] = bias[i];
#pragma unroll
        for (int kc = 0; kc < 8; ++kc) {
            v16h bf = load_b(xnT + ((size_t)(b * TT + t0 + j * 16)) * CCH + kc * 32, CCH);
            acc = wmma16(af[kc], bf, acc);
        }
        const int t = t0 + j * 16 + m;
        if (sec < 2) {
            union { uint4 u; _Float16 h[8]; } pk;
#pragma unroll
            for (int i = 0; i < 8; ++i) pk.h[i] = (_Float16)(acc[i] * QK_SCALE);
            _Float16* dst = (sec == 0 ? qT : kT) + ((size_t)(bh * TT + t)) * CH + r0;
            *(uint4*)dst = pk.u;
        } else {
#pragma unroll
            for (int i = 0; i < 8; ++i)
                vbuf[((size_t)(bh * CH + r0 + i)) * TT + t] = (_Float16)acc[i];
        }
    }
}

// ---------------- K4: flash attention, K/V block-staged in LDS ---------------
__global__ void k_attn(const _Float16* __restrict__ qT,
                       const _Float16* __restrict__ kT,
                       const _Float16* __restrict__ vbuf,
                       _Float16* __restrict__ aT) {
    __shared__ __align__(16) _Float16 Klds[64 * LPITCH];   // [s_local][c]
    __shared__ __align__(16) _Float16 Vlds[64 * LPITCH];   // [c_local][s]
    __shared__ __align__(16) _Float16 plds[8][16 * LPITCH];
    const int warp = threadIdx.x >> 5;
    const int lane = threadIdx.x & 31;
    const int m    = lane & 15;
    const int sel  = lane >> 4;
    const int bh   = blockIdx.x >> 5;        // 512 blocks: 16 bh x 32 row-blocks
    const int rb   = blockIdx.x & 31;
    const int t0   = rb * 128 + warp * 16;
    const int b    = bh >> 2;
    const int h    = bh & 3;

    v16h qA[2];
#pragma unroll
    for (int kc = 0; kc < 2; ++kc)
        qA[kc] = load_a(qT + ((size_t)(bh * TT + t0)) * CH + kc * 32, CH);

    v8f O[4];
    float rmax[8], rsum[8];
#pragma unroll
    for (int j = 0; j < 4; ++j)
#pragma unroll
        for (int i = 0; i < 8; ++i) O[j][i] = 0.f;
#pragma unroll
    for (int i = 0; i < 8; ++i) { rmax[i] = -1e30f; rsum[i] = 0.f; }

    _Float16* pl = plds[warp];

    for (int s0 = 0; s0 < TT; s0 += 64) {
        // ---- cooperative stage: K tile (64s x 64c), V tile (64c x 64s) ----
        __syncthreads();   // previous iteration's consumers done
#pragma unroll
        for (int r2 = 0; r2 < 2; ++r2) {
            const int flat = threadIdx.x + r2 * 256;   // 0..511
            const int row  = flat >> 3;                // 0..63
            const int seg  = flat & 7;                 // 8 f16 segments
            *(uint4*)&Klds[row * LPITCH + seg * 8] =
                *(const uint4*)(kT + ((size_t)(bh * TT + s0 + row)) * CH + seg * 8);
            *(uint4*)&Vlds[row * LPITCH + seg * 8] =
                *(const uint4*)(vbuf + ((size_t)(bh * CH + row)) * TT + s0 + seg * 8);
        }
        __syncthreads();

        // ---- S = q^T k  (4 subtiles of 16 cols, K=64) ----
        v8f S[4];
#pragma unroll
        for (int n = 0; n < 4; ++n) {
            v8f z;
#pragma unroll
            for (int i = 0; i < 8; ++i) z[i] = 0.f;
            v16h kb0 = load_b(&Klds[(n * 16) * LPITCH + 0],  LPITCH);
            v16h kb1 = load_b(&Klds[(n * 16) * LPITCH + 32], LPITCH);
            z    = wmma16(qA[0], kb0, z);
            S[n] = wmma16(qA[1], kb1, z);
        }
        // ---- online softmax: tile row max ----
        float tmax[8];
#pragma unroll
        for (int i = 0; i < 8; ++i) {
            float v = S[0][i];
            v = fmaxf(v, S[1][i]); v = fmaxf(v, S[2][i]); v = fmaxf(v, S[3][i]);
            tmax[i] = v;
        }
#pragma unroll
        for (int off = 1; off < 16; off <<= 1)
#pragma unroll
            for (int i = 0; i < 8; ++i)
                tmax[i] = fmaxf(tmax[i], __shfl_xor(tmax[i], off, 32));

        float alpha[8], nmax[8];
#pragma unroll
        for (int i = 0; i < 8; ++i) {
            nmax[i]  = fmaxf(rmax[i], tmax[i]);
            alpha[i] = exp2f((rmax[i] - nmax[i]) * LOG2E);
            rmax[i]  = nmax[i];
            rsum[i] *= alpha[i];
        }
#pragma unroll
        for (int j = 0; j < 4; ++j)
#pragma unroll
            for (int i = 0; i < 8; ++i) O[j][i] *= alpha[i];

        // ---- P = exp(S - m), row sums, stage into per-wave LDS ----
        float tsum[8];
#pragma unroll
        for (int i = 0; i < 8; ++i) tsum[i] = 0.f;
#pragma unroll
        for (int n = 0; n < 4; ++n)
#pragma unroll
            for (int i = 0; i < 8; ++i) {
                float p = exp2f((S[n][i] - nmax[i]) * LOG2E);
                tsum[i] += p;
                pl[(sel * 8 + i) * LPITCH + n * 16 + m] = (_Float16)p;
            }
#pragma unroll
        for (int off = 1; off < 16; off <<= 1)
#pragma unroll
            for (int i = 0; i < 8; ++i)
                tsum[i] += __shfl_xor(tsum[i], off, 32);
#pragma unroll
        for (int i = 0; i < 8; ++i) rsum[i] += tsum[i];

        // ---- O += P * v^T ----
        v16h pA[2];
#pragma unroll
        for (int kc = 0; kc < 2; ++kc)
            pA[kc] = load_a(pl + kc * 32, LPITCH);
#pragma unroll
        for (int j = 0; j < 4; ++j) {
#pragma unroll
            for (int kc = 0; kc < 2; ++kc) {
                v16h vB = load_b(&Vlds[(j * 16) * LPITCH + kc * 32], LPITCH);
                O[j] = wmma16(pA[kc], vB, O[j]);
            }
        }
    }

    // ---- finalize: divide by row sum, store aT[b][t][h*64 + c] ----
    float rinv[8];
#pragma unroll
    for (int i = 0; i < 8; ++i) rinv[i] = 1.f / rsum[i];
#pragma unroll
    for (int j = 0; j < 4; ++j)
#pragma unroll
        for (int i = 0; i < 8; ++i) {
            const int t = t0 + sel * 8 + i;
            aT[((size_t)(b * TT + t)) * CCH + h * CH + j * 16 + m] =
                (_Float16)(O[j][i] * rinv[i]);
        }
}

// ---------------- K5: proj GEMM + bias + residual (f32 out) ------------------
__global__ void k_proj(const _Float16* __restrict__ wproj,
                       const float* __restrict__ proj_b,
                       const _Float16* __restrict__ aT,
                       const float* __restrict__ x,
                       float* __restrict__ out) {
    const int warp = threadIdx.x >> 5;
    const int strip = blockIdx.x * 8 + warp;        // 4096 strips
    const int b   = strip / 1024;
    const int rem = strip % 1024;
    const int o0  = (rem / 64) * 16;
    const int t0  = (rem % 64) * 64;
    const int lane = threadIdx.x & 31;
    const int m   = lane & 15;
    const int sel = lane >> 4;

    v16h af[8];
#pragma unroll
    for (int kc = 0; kc < 8; ++kc)
        af[kc] = load_a(wproj + o0 * CCH + kc * 32, CCH);

    float bias[8];
#pragma unroll
    for (int i = 0; i < 8; ++i) bias[i] = proj_b[o0 + sel * 8 + i];

#pragma unroll
    for (int j = 0; j < 4; ++j) {
        v8f acc;
#pragma unroll
        for (int i = 0; i < 8; ++i) acc[i] = bias[i];
#pragma unroll
        for (int kc = 0; kc < 8; ++kc) {
            v16h bf = load_b(aT + ((size_t)(b * TT + t0 + j * 16)) * CCH + kc * 32, CCH);
            acc = wmma16(af[kc], bf, acc);
        }
        const int t = t0 + j * 16 + m;
#pragma unroll
        for (int i = 0; i < 8; ++i) {
            const size_t idx = ((size_t)(b * CCH + o0 + sel * 8 + i)) * TT + t;
            out[idx] = x[idx] + acc[i];
        }
    }
}

// ---------------------------------------------------------------------------
extern "C" void kernel_launch(void* const* d_in, const int* in_sizes, int n_in,
                              void* d_out, int out_size, void* d_ws, size_t ws_size,
                              hipStream_t stream) {
    const float* x      = (const float*)d_in[0];
    const float* norm_w = (const float*)d_in[1];
    const float* norm_b = (const float*)d_in[2];
    const float* qkv_w  = (const float*)d_in[3];
    const float* qkv_b  = (const float*)d_in[4];
    const float* proj_w = (const float*)d_in[5];
    const float* proj_b = (const float*)d_in[6];
    float* out = (float*)d_out;

    char* ws = (char*)d_ws;
    float*    stats = (float*)ws;                           // 1 KB
    _Float16* base  = (_Float16*)(ws + 4096);
    const size_t NBT = (size_t)BATCH * TT * CCH;            // 4,194,304
    _Float16* xnT   = base;
    _Float16* qT    = xnT + NBT;
    _Float16* kT    = qT  + NBT;
    _Float16* vbuf  = kT  + NBT;
    _Float16* aT    = vbuf + NBT;
    _Float16* wqkv  = aT  + NBT;
    _Float16* wproj = wqkv + 768 * 256;

    k_cvt_w  <<<1024, 256, 0, stream>>>(qkv_w, proj_w, wqkv, wproj);
    k_gn_stats<<<BATCH * NGROUPS, 256, 0, stream>>>(x, stats);
    k_gn_apply<<<4096, 256, 0, stream>>>(x, norm_w, norm_b, stats, xnT);
    k_qkv    <<<1536, 256, 0, stream>>>(wqkv, qkv_b, xnT, qT, kT, vbuf);
    k_attn   <<<512,  256, 0, stream>>>(qT, kT, vbuf, aT);
    k_proj   <<<512,  256, 0, stream>>>(wproj, proj_b, aT, x, out);
}
